// LocalAttentionLayer_14465449853162
// MI455X (gfx1250) — matrix operand order repaired and checked by reading, not compiled, not run on hardware
//
#include <hip/hip_runtime.h>
#include <hip/hip_bf16.h>

// ---------------------------------------------------------------------------
// LocalAttentionLayer for MI455X (gfx1250, wave32).
//   K0a: resolve same_WqWk, pre-convert Wq/Wk -> bf16, stage biases.
//   K0b: pre-convert key_features -> bf16 (serves K-proj staging and V).
//   K1 : Q/K 1x1-conv projections via v_wmma_f32_16x16x32_bf16 -> bf16 ws.
//   K2 : fused banded logits (WMMA) -> softmax(49) -> banded output (WMMA),
//        absolute-column band trick (WMMA K-dim = columns 0..31, no clamps).
// Workspace: 64MB Qb + 64MB Kb + 32MB Vbf + 128KB Wb + 2KB Bb.
// ---------------------------------------------------------------------------

typedef __bf16 bf16_t;
typedef __attribute__((ext_vector_type(4)))  __bf16 v4bf;
typedef __attribute__((ext_vector_type(8)))  __bf16 v8bf;
typedef __attribute__((ext_vector_type(16))) __bf16 v16bf;
typedef __attribute__((ext_vector_type(8)))  float  v8f;
typedef __attribute__((ext_vector_type(4)))  float  v4f;

#define B_    128
#define CIN_  128
#define CH_   256
#define S_    32
#define NPIX  (S_ * S_)     // 1024
#define KS    7
#define OFF_  3
#define WIN   49

static __device__ __forceinline__ v16bf cat8(v8bf lo, v8bf hi) {
  return __builtin_shufflevector(lo, hi, 0,1,2,3,4,5,6,7,8,9,10,11,12,13,14,15);
}
static __device__ __forceinline__ v8f vzero8() {
  v8f v;
#pragma unroll
  for (int i = 0; i < 8; ++i) v[i] = 0.0f;
  return v;
}
static __device__ __forceinline__ v8bf vzero8bf() {
  v8bf v;
#pragma unroll
  for (int i = 0; i < 8; ++i) v[i] = (bf16_t)0.0f;
  return v;
}
static __device__ __forceinline__ v8f wmma_bf16(v16bf a, v16bf b, v8f c) {
  return __builtin_amdgcn_wmma_f32_16x16x32_bf16(
      false, a, false, b, (short)0, c, false, false);
}

// ---------------------------------------------------------------------------
// Kernel 0a: W/bias prep. Wb[0]=bf16(Wq), Wb[1]=bf16(same?Wq:Wk); Bb likewise.
// ---------------------------------------------------------------------------
__global__ void __launch_bounds__(256)
prep_kernel(const float* __restrict__ Wq, const float* __restrict__ bq,
            const float* __restrict__ Wk, const float* __restrict__ bk,
            const int* __restrict__ sameWW,
            bf16_t* __restrict__ Wb, float* __restrict__ Bb) {
  const bool same = (sameWW[0] != 0);
  const int i = blockIdx.x * 256 + threadIdx.x;      // 0 .. CH_*CIN_-1
  Wb[i]              = (bf16_t)Wq[i];
  Wb[CH_ * CIN_ + i] = (bf16_t)(same ? Wq[i] : Wk[i]);
  if (i < CH_) {
    Bb[i]       = bq[i];
    Bb[CH_ + i] = same ? bq[i] : bk[i];
  }
}

// ---------------------------------------------------------------------------
// Kernel 0b: key_features f32 -> bf16 copy (vectorized, 4 elems/thread).
// ---------------------------------------------------------------------------
__global__ void __launch_bounds__(256)
cvt_kernel(const float* __restrict__ src, bf16_t* __restrict__ dst) {
  const size_t i = ((size_t)blockIdx.x * 256 + threadIdx.x) * 4;
  const v4f f = *(const v4f*)(src + i);
  v4bf o;
#pragma unroll
  for (int e = 0; e < 4; ++e) o[e] = (bf16_t)f[e];
  *(v4bf*)(dst + i) = o;
}

// ---------------------------------------------------------------------------
// Kernel 1: Q/K projections. Block = 8 waves = one (batch, 16-pixel tile);
// 32 wave-tasks (16 o-tiles x {Q,K}), 4 per wave, 4 WMMAs each over CIN.
// Output bf16, layout [b][pixel][channel].
// ---------------------------------------------------------------------------
__global__ void __launch_bounds__(256)
qk_proj_kernel(const float* __restrict__ Xq, const bf16_t* __restrict__ Xkb,
               const bf16_t* __restrict__ Wb, const float* __restrict__ Bb,
               bf16_t* __restrict__ Qb, bf16_t* __restrict__ Kb) {
  const int XP = 136;                                // padded LDS row stride
  __shared__ __attribute__((aligned(16))) bf16_t lq[16 * XP];
  __shared__ __attribute__((aligned(16))) bf16_t lk[16 * XP];

  const int blk = blockIdx.x;
  const int b   = blk >> 6;
  const int p0  = (blk & 63) << 4;
  const int tid = threadIdx.x;

  // Stage X tiles (16 pixels x 128 ch) as bf16 [p][c]; coalesced over pixels.
  for (int i = tid; i < 16 * 128; i += 256) {
    const int c = i >> 4, p = i & 15;
    lq[p * XP + c] = (bf16_t)Xq[(size_t)(b * CIN_ + c) * NPIX + p0 + p];
    lk[p * XP + c] = Xkb[(size_t)(b * CIN_ + c) * NPIX + p0 + p];  // no cvt
  }
  __syncthreads();

  const int lane = tid & 31;
  const int wid  = tid >> 5;
  const int half = lane >> 4;
  const int lm   = lane & 15;

#pragma unroll
  for (int task = 0; task < 4; ++task) {
    const int id  = wid * 4 + task;                  // 0..31
    const int isK = id >> 4;
    const int o0  = (id & 15) << 4;
    const bf16_t* W    = Wb + (size_t)isK * CH_ * CIN_;
    const float*  bias = Bb + isK * CH_;
    const bf16_t* xl   = isK ? lk : lq;

    v8f acc = vzero8();
#pragma unroll
    for (int kb = 0; kb < CIN_; kb += 32) {
      // A = W[o0+lm][.]: elems 0..7 -> c=kb+8*half+e; 8..15 -> c=kb+16+8*half+e
      const bf16_t* wrow = W + (size_t)(o0 + lm) * CIN_ + kb + 8 * half;
      v8bf alo = *(const v8bf*)(wrow);
      v8bf ahi = *(const v8bf*)(wrow + 16);
      // B = X[c][p=lm]: elems 0..7 -> c=kb+8*half+e; 8..15 -> c=kb+16+8*half+e
      const bf16_t* xr = xl + lm * XP + kb + 8 * half;
      v8bf blo = *(const v8bf*)(xr);
      v8bf bhi = *(const v8bf*)(xr + 16);
      acc = wmma_bf16(cat8(alo, ahi), cat8(blo, bhi), acc);
    }

    // D: lane holds pixel col n=lm, rows o = o0 + 8*half + r.
    const float* bptr = bias + o0 + 8 * half;
    v8bf ov;
#pragma unroll
    for (int r = 0; r < 8; ++r) ov[r] = (bf16_t)(acc[r] + bptr[r]);
    bf16_t* dst = (isK ? Kb : Qb) +
                  ((size_t)(b * NPIX + p0 + lm)) * CH_ + o0 + 8 * half;
    *(v8bf*)dst = ov;
  }
}

// ---------------------------------------------------------------------------
// Kernel 2: fused banded attention; one wave per (b, row h, 16-pixel tile).
// ---------------------------------------------------------------------------
__global__ void __launch_bounds__(256)
attn_kernel(const bf16_t* __restrict__ Qb, const bf16_t* __restrict__ Kb,
            const bf16_t* __restrict__ Vb, float* __restrict__ out) {
  __shared__ __attribute__((aligned(16))) float  bandLds[8][2][16 * 17];
  __shared__ __attribute__((aligned(16))) float  attnLds[8][16][52];
  __shared__ __attribute__((aligned(16))) bf16_t aRowLds[8][16][KS][32];

  const int tid  = threadIdx.x;
  const int lane = tid & 31;
  const int wid  = tid >> 5;
  const int half = lane >> 4;
  const int lm   = lane & 15;

  const int gw = blockIdx.x * 8 + wid;               // 0..8191 tile id
  const int b  = gw >> 6;
  const int r  = gw & 63;
  const int h  = r >> 1;
  const int w0 = (r & 1) << 4;

  // Prefetch the 7 bf16 V rows (row = 32 bf16 = 64B; 4 channels/lane pass).
#pragma unroll
  for (int dh = 0; dh < KS; ++dh) {
    const int hc = min(max(h - OFF_ + dh, 0), S_ - 1);
#pragma unroll
    for (int i = 0; i < 4; ++i) {
      const int c = lane + 32 * i;
      __builtin_prefetch(Vb + ((size_t)(b * CIN_ + c)) * NPIX + hc * S_, 0, 1);
    }
  }

  // ---- Q A-operand: 16 pixels x 256 ch, 8 k-chunks of 32 ----
  v16bf aQ[8];
  const bf16_t* qrow = Qb + ((size_t)(b * NPIX + h * S_ + w0 + lm)) * CH_;
#pragma unroll
  for (int kc = 0; kc < 8; ++kc) {
    const bf16_t* p = qrow + kc * 32 + 8 * half;
    v8bf lo = *(const v8bf*)(p);
    v8bf hi = *(const v8bf*)(p + 16);
    aQ[kc] = cat8(lo, hi);
  }

  // ---- Logits: per neighbor row, Q(16x256) x K^T for absolute cols 0..31 ----
  for (int dh = 0; dh < KS; ++dh) {
    const int  hr  = h - OFF_ + dh;
    const bool hok = (hr >= 0) && (hr < S_);
    const int  hc  = min(max(hr, 0), S_ - 1);        // clamp; masked below

    v8f acc0 = vzero8(), acc1 = vzero8();
#pragma unroll
    for (int t = 0; t < 2; ++t) {
      const bf16_t* krow = Kb + ((size_t)(b * NPIX + hc * S_ + t * 16 + lm)) * CH_;
      v8f acc = vzero8();
#pragma unroll
      for (int kc = 0; kc < 8; ++kc) {
        const bf16_t* p = krow + kc * 32 + 8 * half;
        v8bf lo = *(const v8bf*)(p);
        v8bf hi = *(const v8bf*)(p + 16);
        acc = wmma_bf16(aQ[kc], cat8(lo, hi), acc);
      }
      if (t == 0) acc0 = acc; else acc1 = acc;
    }

#pragma unroll
    for (int rr = 0; rr < 8; ++rr) {
      bandLds[wid][0][(8 * half + rr) * 17 + lm] = acc0[rr];
      bandLds[wid][1][(8 * half + rr) * 17 + lm] = acc1[rr];
    }
    __syncthreads();

    if (lane < 16) {                                  // extract 7 taps per pixel
#pragma unroll
      for (int dw = 0; dw < KS; ++dw) {
        const int  j  = w0 + lm - OFF_ + dw;          // absolute column
        const bool ok = hok && (j >= 0) && (j < S_);
        const int  jc = min(max(j, 0), S_ - 1);
        const float v = bandLds[wid][jc >> 4][lm * 17 + (jc & 15)];
        attnLds[wid][lm][dh * KS + dw] = ok ? v : -1e30f;
      }
    }
    __syncthreads();
  }

  // ---- Softmax over 49 + build banded bf16 A' rows (abs cols 0..31) ----
  if (lane < 16) {
    float* a = attnLds[wid][lm];
    float mx = -1e30f;
#pragma unroll
    for (int i = 0; i < WIN; ++i) mx = fmaxf(mx, a[i]);
    float s = 0.0f;
#pragma unroll
    for (int i = 0; i < WIN; ++i) { float e = __expf(a[i] - mx); a[i] = e; s += e; }
    const float inv = 1.0f / s;                       // center tap always valid

    const int k0 = w0 + lm - OFF_;                    // band start column
#pragma unroll
    for (int dh = 0; dh < KS; ++dh) {
      bf16_t* row = &aRowLds[wid][lm][dh][0];
      v8bf z = vzero8bf();
#pragma unroll
      for (int k4 = 0; k4 < 4; ++k4) *(v8bf*)(row + k4 * 8) = z;
#pragma unroll
      for (int dw = 0; dw < KS; ++dw) {
        const int k = k0 + dw;
        if (k >= 0 && k < S_) row[k] = (bf16_t)(a[dh * KS + dw] * inv);
      }
    }
  }
  __syncthreads();

  // ---- Output: out[m][c] += A'(16x32 band) x V_row(32 cols x 16 ch) ----
  v8f accO[8];
#pragma unroll
  for (int i = 0; i < 8; ++i) accO[i] = vzero8();

  for (int dh = 0; dh < KS; ++dh) {
    const int hc = min(max(h - OFF_ + dh, 0), S_ - 1); // masked taps are 0

    const bf16_t* arow = &aRowLds[wid][lm][dh][0];
    v8bf alo = *(const v8bf*)(arow + 8 * half);
    v8bf ahi = *(const v8bf*)(arow + 16 + 8 * half);
    const v16bf aA = cat8(alo, ahi);

#pragma unroll
    for (int nc = 0; nc < 8; ++nc) {
      // B'[k=col][n=channel]: lane lm = channel nc*16+lm; cols 8*half..+7 and
      // 16+8*half..+7 -> two aligned 16B loads, zero converts.
      const bf16_t* vrow = Vb + ((size_t)(b * CIN_ + nc * 16 + lm)) * NPIX +
                           hc * S_ + 8 * half;
      v8bf blo = *(const v8bf*)(vrow);
      v8bf bhi = *(const v8bf*)(vrow + 16);
      accO[nc] = wmma_bf16(aA, cat8(blo, bhi), accO[nc]);
    }
  }

  // D: lane holds channel col n=nc*16+lm, pixels m=8*half..+7 (w-contiguous).
#pragma unroll
  for (int nc = 0; nc < 8; ++nc) {
    float* dst = out + ((size_t)(b * CIN_ + nc * 16 + lm)) * NPIX +
                 h * S_ + w0 + 8 * half;
    v4f s0, s1;
#pragma unroll
    for (int rr = 0; rr < 4; ++rr) { s0[rr] = accO[nc][rr]; s1[rr] = accO[nc][4 + rr]; }
    *(v4f*)(dst)     = s0;
    *(v4f*)(dst + 4) = s1;
  }
}

// ---------------------------------------------------------------------------
extern "C" void kernel_launch(void* const* d_in, const int* in_sizes, int n_in,
                              void* d_out, int out_size, void* d_ws, size_t ws_size,
                              hipStream_t stream) {
  (void)in_sizes; (void)n_in; (void)out_size; (void)ws_size;
  const float* Xq = (const float*)d_in[0];   // query_features [B,CIN,S,S]
  const float* Xk = (const float*)d_in[1];   // key_features   [B,CIN,S,S]
  const float* Wq = (const float*)d_in[2];   // [CH,CIN]
  const float* bq = (const float*)d_in[3];   // [CH]
  const float* Wk = (const float*)d_in[4];   // [CH,CIN]
  const float* bk = (const float*)d_in[5];   // [CH]
  // d_in[6] = vis_CA (unused by the math)
  const int* same = (const int*)d_in[7];     // same_WqWk flag

  const size_t NQK = (size_t)B_ * NPIX * CH_;          // 33,554,432 elems
  const size_t NX  = (size_t)B_ * NPIX * CIN_;         // 16,777,216 elems
  bf16_t* Qb  = (bf16_t*)d_ws;                         // 64 MB
  bf16_t* Kb  = Qb + NQK;                              // 64 MB
  bf16_t* Vbf = Kb + NQK;                              // 32 MB
  bf16_t* Wb  = Vbf + NX;                              // 128 KB (2 matrices)
  float*  Bb  = (float*)(Wb + 2 * CH_ * CIN_);         // 2 KB  (2 bias vecs)

  prep_kernel<<<(CH_ * CIN_) / 256, 256, 0, stream>>>(Wq, bq, Wk, bk, same, Wb, Bb);
  cvt_kernel<<<(int)(NX / (256 * 4)), 256, 0, stream>>>(Xk, Vbf);
  qk_proj_kernel<<<B_ * 64, 256, 0, stream>>>(Xq, Vbf, Wb, Bb, Qb, Kb);
  attn_kernel<<<(B_ * 64) / 8, 256, 0, stream>>>(Qb, Kb, Vbf, (float*)d_out);
}
